// CorefModel_5987184410917
// MI455X (gfx1250) — matrix operand order repaired and checked by reading, not compiled, not run on hardware
//
// CorefModel pipeline for MI455X (gfx1250):
//  - bf16 WMMA GEMMs (v_wmma_f32_16x16x32_bf16), 128x128x32 tiles, 8 wave32 waves
//  - async global->LDS staging (global_load_async_to_lds_b128 + s_wait_asynccnt)
//  - one-time bf16 weight transpose/convert pre-pass (all fp32->bf16 out of hot loop)
//  - atomic segment-softmax GAT scatter
#include <hip/hip_runtime.h>
#include <hip/hip_bf16.h>

#define NT_   8192
#define NC_   16384
#define H_    8
#define DTOK  768
#define DCH_  128
#define DD    1664      // 2*DTOK + DCH
#define ECC   65536
#define ECT   131072
#define HID   1024

typedef unsigned short u16;
typedef __attribute__((ext_vector_type(16))) __bf16 bfx16;
typedef __attribute__((ext_vector_type(8)))  float  fx8;

union FragU { uint4 q[2]; bfx16 v; };

__device__ __forceinline__ u16 f2bf(float f) {
  unsigned u = __float_as_uint(f);
  u += 0x7fffu + ((u >> 16) & 1u);           // round-to-nearest-even
  return (u16)(u >> 16);
}
// order-preserving float<->u32 key for atomicMax-based segment max
__device__ __forceinline__ unsigned encf(float x) {
  unsigned u = __float_as_uint(x);
  return (u & 0x80000000u) ? ~u : (u | 0x80000000u);
}
__device__ __forceinline__ float decf(unsigned k) {
  unsigned u = (k & 0x80000000u) ? (k ^ 0x80000000u) : ~k;
  return __uint_as_float(u);
}

// ---- CDNA5 async copy helpers (ISA 10. async LDS<->memory, ASYNCcnt) -------
__device__ __forceinline__ void async_ld16(unsigned lds_byte_off, const void* g) {
  asm volatile("global_load_async_to_lds_b128 %0, %1, off"
               :: "v"(lds_byte_off), "v"(g) : "memory");
}
__device__ __forceinline__ void wait_async4() {   // own-wave ASYNCcnt <= 4
  asm volatile("s_wait_asynccnt 0x4" ::: "memory");
}
__device__ __forceinline__ void wait_async0() {   // own-wave ASYNCcnt == 0
  asm volatile("s_wait_asynccnt 0x0" ::: "memory");
}

// ---------------------------------------------------------------------------
// GEMM: C[M,N] = act(A[M,K] @ B[K,N] + bias).
// A: bf16 row-major [M,K]; Bt: bf16 PRE-TRANSPOSED [N,K]; C: fp32 [M,N].
// 128x128x32 tile, 256 threads = 8 wave32 waves (4 along M x 2 along N),
// each wave owns 2x4 16x16x32 WMMA accumulators. Double-buffered LDS filled
// by async global->LDS b128 copies. All dims are multiples of 128/32.
// ---------------------------------------------------------------------------
__global__ __launch_bounds__(256)
void gemm_bf16_wmma(const u16* __restrict__ A, const u16* __restrict__ Bt,
                    float* __restrict__ C, const float* __restrict__ bias,
                    int M, int N, int K, int act)
{
  __shared__ __align__(16) u16 sA[2][128 * 40];  // [row][k], stride 40 (pad)
  __shared__ __align__(16) u16 sB[2][128 * 40];  // [col][k] (Bt rows)

  const int tid  = threadIdx.x;
  const int lane = tid & 31;
  const int wave = tid >> 5;
  const int wm   = wave & 3;    // 4 waves along M, 32 rows each
  const int wn   = wave >> 2;   // 2 waves along N, 64 cols each
  const int bm   = blockIdx.y * 128;
  const int bn   = blockIdx.x * 128;

  fx8 acc[2][4];
#pragma unroll
  for (int i = 0; i < 2; ++i)
#pragma unroll
    for (int j = 0; j < 4; ++j)
#pragma unroll
      for (int r = 0; r < 8; ++r) acc[i][j][r] = 0.f;

  // per-thread 16B chunk coords: 512 chunks cover a 128x32 bf16 tile
  const int r0 = tid >> 2,           c0 = (tid & 3) << 3;
  const int r1 = (tid + 256) >> 2,   c1 = ((tid + 256) & 3) << 3;

  auto issue = [&](int kt, int buf) {   // 4 async b128 per wave per tile-pair
    const int k0 = kt << 5;
    async_ld16((unsigned)(size_t)&sA[buf][r0 * 40 + c0], A  + (size_t)(bm + r0) * K + k0 + c0);
    async_ld16((unsigned)(size_t)&sB[buf][r0 * 40 + c0], Bt + (size_t)(bn + r0) * K + k0 + c0);
    async_ld16((unsigned)(size_t)&sA[buf][r1 * 40 + c1], A  + (size_t)(bm + r1) * K + k0 + c1);
    async_ld16((unsigned)(size_t)&sB[buf][r1 * 40 + c1], Bt + (size_t)(bn + r1) * K + k0 + c1);
  };

  const int kT = K >> 5;
  issue(0, 0);

  const int mrow = lane & 15;
  const int kbA  = (lane >> 4) << 3;   // A frag: K runs {kb..kb+7, kb+16..kb+23}
  const int kbB  = (lane >> 4) << 3;   // B frag: K runs {kb..kb+7, kb+16..kb+23}

  for (int kt = 0; kt < kT; ++kt) {
    const int buf = kt & 1;
    if (kt + 1 < kT) { issue(kt + 1, buf ^ 1); wait_async4(); }
    else             { wait_async0(); }
    __syncthreads();                       // all waves' tile-kt asyncs visible

    FragU fbf[4];
#pragma unroll
    for (int nt = 0; nt < 4; ++nt) {
      const u16* pb = &sB[buf][(wn * 64 + nt * 16 + mrow) * 40 + kbB];
      fbf[nt].q[0] = *(const uint4*)pb;
      fbf[nt].q[1] = *(const uint4*)(pb + 16);
    }
#pragma unroll
    for (int mt = 0; mt < 2; ++mt) {
      FragU fa;
      const u16* pa = &sA[buf][(wm * 32 + mt * 16 + mrow) * 40 + kbA];
      fa.q[0] = *(const uint4*)pa;
      fa.q[1] = *(const uint4*)(pa + 16);
#pragma unroll
      for (int nt = 0; nt < 4; ++nt)
        acc[mt][nt] = __builtin_amdgcn_wmma_f32_16x16x32_bf16(
            false, fa.v, false, fbf[nt].v, (short)0, acc[mt][nt], false, false);
    }
    __syncthreads();                       // everyone done reading buf
  }

  // epilogue: D layout -> row = r + 8*(lane>=16), col = lane&15
#pragma unroll
  for (int mt = 0; mt < 2; ++mt)
#pragma unroll
    for (int nt = 0; nt < 4; ++nt) {
      int n  = bn + wn * 64 + nt * 16 + (lane & 15);
      int m0 = bm + wm * 32 + mt * 16 + ((lane >> 4) << 3);
      float bv = act ? bias[n] : 0.f;
#pragma unroll
      for (int r = 0; r < 8; ++r) {
        float v = acc[mt][nt][r] + bv;
        if (act == 1) v = fmaxf(v, 0.f);
        C[(size_t)(m0 + r) * N + n] = v;
      }
    }
}

// ---------------------------------------------------------------------------
// Pre-pass conversions
// ---------------------------------------------------------------------------
// dst[n*K + k] = bf16(src[k*N + n]); K,N multiples of 32
__global__ __launch_bounds__(256) void transpose_cvt_k(u16* __restrict__ dst,
                                                       const float* __restrict__ src,
                                                       int K, int N) {
  __shared__ float t[32][33];
  int k0 = blockIdx.y * 32, n0 = blockIdx.x * 32;
  int tx = threadIdx.x & 31, ty = threadIdx.x >> 5;   // 32x8
#pragma unroll
  for (int i = 0; i < 32; i += 8)
    t[ty + i][tx] = src[(size_t)(k0 + ty + i) * N + (n0 + tx)];
  __syncthreads();
#pragma unroll
  for (int i = 0; i < 32; i += 8)
    dst[(size_t)(n0 + ty + i) * K + (k0 + tx)] = f2bf(t[tx][ty + i]);
}

__global__ void cvt_bf16_k(u16* __restrict__ dst, const float* __restrict__ src, int n4) {
  int i = blockIdx.x * 256 + threadIdx.x;
  if (i >= n4) return;
  float4 v = ((const float4*)src)[i];
  ushort4 u;
  u.x = f2bf(v.x); u.y = f2bf(v.y); u.z = f2bf(v.z); u.w = f2bf(v.w);
  ((ushort4*)dst)[i] = u;
}

// ---------------------------------------------------------------------------
// Elementwise / GAT kernels
// ---------------------------------------------------------------------------
__global__ void fill_f32_k(float* p, float v, int n) {
  int i = blockIdx.x * 256 + threadIdx.x;
  if (i < n) p[i] = v;
}
__global__ void fill_u32_k(unsigned* p, unsigned v, int n) {
  int i = blockIdx.x * 256 + threadIdx.x;
  if (i < n) p[i] = v;
}

__global__ void build_cons_k(u16* __restrict__ cons, const float* __restrict__ toksrc,
                             const float* __restrict__ table, const int* __restrict__ st,
                             const int* __restrict__ en, const int* __restrict__ lb) {
  size_t idx = (size_t)blockIdx.x * 256 + threadIdx.x;
  if (idx >= (size_t)NC_ * DD) return;
  int n = (int)(idx / DD);
  int d = (int)(idx - (size_t)n * DD);
  float v;
  if (d < DTOK)          v = toksrc[(size_t)st[n] * DTOK + d];
  else if (d < 2 * DTOK) v = toksrc[(size_t)en[n] * DTOK + (d - DTOK)];
  else                   v = table[lb[n] * DCH_ + (d - 2 * DTOK)];
  cons[idx] = f2bf(v);
}

// per-(node,head) dot with attention vector: one wave32 per (n,h)
__global__ void head_score_k(const float* __restrict__ h, const float* __restrict__ a,
                             float* __restrict__ out, int nNodes, int dh, int Dtot) {
  int gw   = (int)(((size_t)blockIdx.x * blockDim.x + threadIdx.x) >> 5);
  int lane = threadIdx.x & 31;
  if (gw >= nNodes * H_) return;
  int n = gw >> 3, hh = gw & 7;
  const float* row = h + (size_t)n * Dtot + hh * dh;
  const float* av  = a + hh * dh;
  float s = 0.f;
  for (int d = lane; d < dh; d += 32) s += row[d] * av[d];
#pragma unroll
  for (int off = 16; off > 0; off >>= 1) s += __shfl_xor(s, off, 32);
  if (lane == 0) out[gw] = s;
}

__global__ void edge_logit_k(const float* __restrict__ ss, const float* __restrict__ sd,
                             const int* __restrict__ src, const int* __restrict__ dst,
                             float* __restrict__ elog, unsigned* __restrict__ smax, int E) {
  int i = blockIdx.x * 256 + threadIdx.x;
  if (i >= E * H_) return;
  int e = i >> 3, h = i & 7;
  float l = ss[src[e] * H_ + h] + sd[dst[e] * H_ + h];
  l = (l > 0.f) ? l : 0.2f * l;                   // leaky_relu 0.2
  elog[i] = l;
  atomicMax(smax + dst[e] * H_ + h, encf(l));
}

__global__ void edge_exp_k(float* __restrict__ elog, const int* __restrict__ dst,
                           const unsigned* __restrict__ smax, float* __restrict__ ssum, int E) {
  int i = blockIdx.x * 256 + threadIdx.x;
  if (i >= E * H_) return;
  int e = i >> 3, h = i & 7;
  float m  = decf(smax[dst[e] * H_ + h]);
  float ev = __expf(elog[i] - m);
  elog[i] = ev;
  atomicAdd(ssum + dst[e] * H_ + h, ev);
}

// payload scatter: out[dst, :] += alpha(e,h) * hs[src, :]
template <int DH, int DTOT>
__global__ void gat_scatter_k(const float* __restrict__ hs, const float* __restrict__ elog,
                              const float* __restrict__ ssum, const int* __restrict__ src,
                              const int* __restrict__ dst, float* __restrict__ out) {
  int e = blockIdx.x;
  int d = blockIdx.y * 128 + threadIdx.x;
  int s = src[e], t = dst[e];
  int h = d / DH;
  float a = elog[e * H_ + h] / (ssum[t * H_ + h] + 1e-9f);
  atomicAdd(out + (size_t)t * DTOT + d, a * hs[(size_t)s * DTOT + d]);
}

// scores[n,j] = relu-hidden[n,:] . w2   (one wave per row)
__global__ void hidden_dot_k(const float* __restrict__ hid, const float* __restrict__ w2,
                             float* __restrict__ scores, int rows, int j) {
  int gw   = (int)(((size_t)blockIdx.x * blockDim.x + threadIdx.x) >> 5);
  int lane = threadIdx.x & 31;
  if (gw >= rows) return;
  const float* r = hid + (size_t)gw * HID;
  float s = 0.f;
  for (int k = lane; k < HID; k += 32) s += r[k] * w2[k];
#pragma unroll
  for (int off = 16; off > 0; off >>= 1) s += __shfl_xor(s, off, 32);
  if (lane == 0) scores[gw * 2 + j] = s;
}

// cons_fused (bf16, GEMM operand) = softmax(scores) . {order0, order1}
__global__ void fuse_orders_k(u16* __restrict__ outc, const float* __restrict__ o0,
                              const float* __restrict__ o1, const float* __restrict__ sc) {
  size_t idx = (size_t)blockIdx.x * 256 + threadIdx.x;
  if (idx >= (size_t)NC_ * DD) return;
  int n = (int)(idx / DD);
  float s0 = sc[n * 2], s1 = sc[n * 2 + 1];
  float m  = fmaxf(s0, s1);
  float e0 = __expf(s0 - m), e1 = __expf(s1 - m);
  float w0 = e0 / (e0 + e1);
  outc[idx] = f2bf(w0 * o0[idx] + (1.f - w0) * o1[idx]);
}

__global__ void concat_k(u16* __restrict__ cat, const float* __restrict__ tok,
                         const float* __restrict__ tokc) {
  size_t idx = (size_t)blockIdx.x * 256 + threadIdx.x;
  if (idx >= (size_t)NT_ * 2 * DTOK) return;
  int n = (int)(idx / (2 * DTOK));
  int d = (int)(idx - (size_t)n * 2 * DTOK);
  float v = (d < DTOK) ? tok[(size_t)n * DTOK + d] : tokc[(size_t)n * DTOK + d - DTOK];
  cat[idx] = f2bf(v);
}

__global__ void final_mix_k(float* __restrict__ out, const float* __restrict__ tok,
                            const float* __restrict__ tokc, const float* __restrict__ gate) {
  size_t idx = (size_t)blockIdx.x * 256 + threadIdx.x;
  if (idx >= (size_t)NT_ * DTOK) return;
  float g = 1.f / (1.f + __expf(-gate[idx]));
  out[idx] = g * tok[idx] + (1.f - g) * tokc[idx];
}

// ---------------------------------------------------------------------------
extern "C" void kernel_launch(void* const* d_in, const int* in_sizes, int n_in,
                              void* d_out, int out_size, void* d_ws, size_t ws_size,
                              hipStream_t stream) {
  const float* tok    = (const float*)d_in[0];
  const float* table  = (const float*)d_in[1];
  const float* Wf     = (const float*)d_in[2];
  const float* af_s   = (const float*)d_in[3];
  const float* af_d   = (const float*)d_in[4];
  const float* Wb     = (const float*)d_in[5];
  const float* ab_s   = (const float*)d_in[6];
  const float* ab_d   = (const float*)d_in[7];
  const float* Wk     = (const float*)d_in[8];
  const float* Wq     = (const float*)d_in[9];
  const float* act_s  = (const float*)d_in[10];
  const float* act_d  = (const float*)d_in[11];
  const float* w1     = (const float*)d_in[12];
  const float* b1     = (const float*)d_in[13];
  const float* w2     = (const float*)d_in[14];
  const float* fw     = (const float*)d_in[15];
  const float* fb     = (const float*)d_in[16];
  const int*   starts = (const int*)d_in[17];
  const int*   ends   = (const int*)d_in[18];
  const int*   labels = (const int*)d_in[19];
  const int*   cc_src = (const int*)d_in[20];
  const int*   cc_dst = (const int*)d_in[21];
  const int*   ct_src = (const int*)d_in[22];
  const int*   ct_dst = (const int*)d_in[23];

  char* base = (char*)d_ws;
  size_t off = 0;
  auto alloc = [&](size_t bytes) -> void* {
    void* p = base + off;
    off += (bytes + 255) & ~(size_t)255;
    return p;
  };
  // fp32 buffers
  float*    hf     = (float*)alloc((size_t)NC_ * DD * 4);
  float*    hb     = (float*)alloc((size_t)NC_ * DD * 4);
  float*    order0 = (float*)alloc((size_t)NC_ * DD * 4);
  float*    order1 = (float*)alloc((size_t)NC_ * DD * 4);
  float*    hidden = (float*)alloc((size_t)NC_ * HID * 4);
  float*    tokc   = (float*)alloc((size_t)NT_ * DTOK * 4);
  float*    scores = (float*)alloc((size_t)NC_ * 2 * 4);
  float*    sSa    = (float*)alloc((size_t)NC_ * H_ * 4);
  float*    sSb    = (float*)alloc((size_t)NC_ * H_ * 4);
  float*    sSc    = (float*)alloc((size_t)NC_ * H_ * 4);
  float*    sSd    = (float*)alloc((size_t)NC_ * H_ * 4);
  float*    sQ     = (float*)alloc((size_t)NT_ * H_ * 4);
  unsigned* smax   = (unsigned*)alloc((size_t)NC_ * H_ * 4);
  float*    ssum   = (float*)alloc((size_t)NC_ * H_ * 4);
  float*    elog   = (float*)alloc((size_t)ECT * H_ * 4);
  // bf16 buffers
  u16* consB = (u16*)alloc((size_t)NC_ * DD * 2);         // cons / cons_fused
  u16* ordB  = (u16*)alloc((size_t)NC_ * DD * 2);         // current order, bf16
  u16* tokqB = (u16*)alloc((size_t)NT_ * DTOK * 2);       // GEMM-A for Wq
  u16* catB  = (u16*)alloc((size_t)NT_ * 2 * DTOK * 2);   // [tok|tokc] bf16
  u16* WfT = (u16*)alloc((size_t)DD * DD * 2);            // weights, [N,K] bf16
  u16* WbT = (u16*)alloc((size_t)DD * DD * 2);
  u16* w1T = (u16*)alloc((size_t)HID * DD * 2);
  u16* WkT = (u16*)alloc((size_t)DTOK * DD * 2);
  u16* WqT = (u16*)alloc((size_t)DTOK * DTOK * 2);
  u16* fwT = (u16*)alloc((size_t)DTOK * 2 * DTOK * 2);
  // aliases (lifetimes disjoint on the single stream):
  float* hk   = hidden;   // [NC,768]  <= [NC,1024]; hidden done before Wk GEMM
  float* hq   = hf;       // [NT,768]  <= hf free after j-loop scatters
  float* gate = order1;   // [NT,768]  used only after the iteration loop

  auto gemm = [&](const u16* A, const u16* Bt, float* C, const float* bias,
                  int M, int N, int K, int act) {
    dim3 g(N / 128, M / 128);
    gemm_bf16_wmma<<<g, 256, 0, stream>>>(A, Bt, C, bias, M, N, K, act);
  };
  auto tcvt = [&](u16* dst, const float* src, int K, int N) {
    dim3 g(N / 32, K / 32);
    transpose_cvt_k<<<g, 256, 0, stream>>>(dst, src, K, N);
  };
  auto cvt = [&](u16* dst, const float* src, size_t n) {
    cvt_bf16_k<<<(unsigned)((n / 4 + 255) / 256), 256, 0, stream>>>(dst, src, (int)(n / 4));
  };
  auto fillf = [&](float* p, size_t n) {
    fill_f32_k<<<(unsigned)((n + 255) / 256), 256, 0, stream>>>(p, 0.f, (int)n);
  };
  auto run_gat = [&](const float* hs, const float* ss, const float* sd,
                     const int* es, const int* ed, int E, int Nd, int Dtot, int dh,
                     float* outAcc) {
    int nh = Nd * H_;
    fill_u32_k<<<(nh + 255) / 256, 256, 0, stream>>>(smax, 0u, nh);
    fill_f32_k<<<(nh + 255) / 256, 256, 0, stream>>>(ssum, 0.f, nh);
    int eh = E * H_;
    edge_logit_k<<<(eh + 255) / 256, 256, 0, stream>>>(ss, sd, es, ed, elog, smax, E);
    edge_exp_k<<<(eh + 255) / 256, 256, 0, stream>>>(elog, ed, smax, ssum, E);
    dim3 g(E, Dtot / 128);
    if (dh == 208) gat_scatter_k<208, DD  ><<<g, 128, 0, stream>>>(hs, elog, ssum, es, ed, outAcc);
    else           gat_scatter_k<96,  DTOK><<<g, 128, 0, stream>>>(hs, elog, ssum, es, ed, outAcc);
  };
  auto hscore = [&](const float* h, const float* a, float* out, int nNodes, int dh, int Dtot) {
    size_t thr = (size_t)nNodes * H_ * 32;
    head_score_k<<<(unsigned)((thr + 255) / 256), 256, 0, stream>>>(h, a, out, nNodes, dh, Dtot);
  };

  // one-time weight transpose+convert (fp32 [K,N] -> bf16 [N,K])
  tcvt(WfT, Wf, DD, DD);
  tcvt(WbT, Wb, DD, DD);
  tcvt(w1T, w1, DD, HID);
  tcvt(WkT, Wk, DD, DTOK);
  tcvt(WqT, Wq, DTOK, DTOK);
  tcvt(fwT, fw, 2 * DTOK, DTOK);

  const float* tcur = tok;
  for (int it = 0; it < 2; ++it) {
    build_cons_k<<<(unsigned)(((size_t)NC_ * DD + 255) / 256), 256, 0, stream>>>(
        consB, tcur, table, starts, ends, labels);
    gemm(consB, WfT, hf, nullptr, NC_, DD, DD, 0);
    gemm(consB, WbT, hb, nullptr, NC_, DD, DD, 0);
    hscore(hf, af_s, sSa, NC_, DD / H_, DD);
    hscore(hf, af_d, sSb, NC_, DD / H_, DD);
    hscore(hb, ab_s, sSc, NC_, DD / H_, DD);
    hscore(hb, ab_d, sSd, NC_, DD / H_, DD);
    for (int j = 0; j < 2; ++j) {
      float* ord = j ? order1 : order0;
      fillf(ord, (size_t)NC_ * DD);
      run_gat(hf, sSa, sSb, cc_src + (size_t)j * ECC,       cc_dst + (size_t)j * ECC,       ECC, NC_, DD, 208, ord);
      run_gat(hb, sSc, sSd, cc_src + (size_t)(2 + j) * ECC, cc_dst + (size_t)(2 + j) * ECC, ECC, NC_, DD, 208, ord);
      cvt(ordB, ord, (size_t)NC_ * DD);
      gemm(ordB, w1T, hidden, b1, NC_, HID, DD, 1);   // relu(x@w1 + b1)
      hidden_dot_k<<<(NC_ * 32 + 255) / 256, 256, 0, stream>>>(hidden, w2, scores, NC_, j);
    }
    fuse_orders_k<<<(unsigned)(((size_t)NC_ * DD + 255) / 256), 256, 0, stream>>>(
        consB, order0, order1, scores);               // consB <- cons_fused (bf16)
    gemm(consB, WkT, hk, nullptr, NC_, DTOK, DD, 0);
    cvt(tokqB, tcur, (size_t)NT_ * DTOK);
    gemm(tokqB, WqT, hq, nullptr, NT_, DTOK, DTOK, 0);
    hscore(hk, act_s, sSa, NC_, DTOK / H_, DTOK);
    hscore(hq, act_d, sQ,  NT_, DTOK / H_, DTOK);
    fillf(tokc, (size_t)NT_ * DTOK);
    run_gat(hk, sSa, sQ, ct_src, ct_dst, ECT, NT_, DTOK, 96, tokc);
    tcur = tokc;
  }

  concat_k<<<(unsigned)(((size_t)NT_ * 2 * DTOK + 255) / 256), 256, 0, stream>>>(catB, tok, tokc);
  gemm(catB, fwT, gate, fb, NT_, DTOK, 2 * DTOK, 2);  // bias only
  final_mix_k<<<(unsigned)(((size_t)NT_ * DTOK + 255) / 256), 256, 0, stream>>>(
      (float*)d_out, tok, tokc, gate);
}
